// Net_22840636080718
// MI455X (gfx1250) — matrix operand order repaired and checked by reading, not compiled, not run on hardware
//
#include <hip/hip_runtime.h>
#include <math.h>

// ---------------------------------------------------------------------------
// MI455X (gfx1250) fused MeshGraphNet forward.
// wave32, WMMA bf16 16x16x32, weights LDS-resident (320KB WGP LDS),
// activations bf16-staged in LDS, f32 accumulation.
// ---------------------------------------------------------------------------

typedef unsigned short ushort_t;
typedef __attribute__((ext_vector_type(16))) __bf16 v16bf;
typedef __attribute__((ext_vector_type(8)))  float  v8f;

union ABfrag { uint4 q[2]; v16bf v; };

__device__ __forceinline__ void cfence() { asm volatile("" ::: "memory"); }

// Hard scheduling-region boundary: forbids the machine scheduler from
// interleaving the batched LDS loads with the WMMA issue group, so the
// 8 B-fragments stay in distinct registers and the matrix pipe gets
// 8 back-to-back v_wmma per k-chunk behind a single s_wait_dscnt.
__device__ __forceinline__ void sched_fence() {
#if __has_builtin(__builtin_amdgcn_sched_barrier)
    __builtin_amdgcn_sched_barrier(0);
#else
    asm volatile("" ::: "memory");
#endif
}

__device__ __forceinline__ ushort_t f2bf(float f) {
    union { float f; unsigned u; } v; v.f = f;
    unsigned u = v.u;
    u = (u + 0x7FFFu + ((u >> 16) & 1u)) >> 16;   // RNE truncate to bf16
    return (ushort_t)u;
}
__device__ __forceinline__ float bf2f(ushort_t h) {
    union { unsigned u; float f; } v; v.u = ((unsigned)h) << 16;
    return v.f;
}
__device__ __forceinline__ void store4bf(ushort_t* p, float4 v) {
    uint2 w;
    w.x = (unsigned)f2bf(v.x) | ((unsigned)f2bf(v.y) << 16);
    w.y = (unsigned)f2bf(v.z) | ((unsigned)f2bf(v.w) << 16);
    *(uint2*)p = w;
}

// ---- WMMA GEMM inner loop: 16(M) x 128(N) tile, K accumulation ------------
// A staged row-major bf16 in LDS (stride sa halves), W staged as Wt[n][k]
// (stride sw halves). Lane mapping (contiguous k order, same for A and B):
//   A frag: m = lane&15, k = (lane>>4)*16 + i                 -> one 32B run
//   B frag: n = lane&15, k = (lane>>4)*16 + i                 -> one 32B run
__device__ __forceinline__ void gemm_accum(const ushort_t* Astg, int sa,
                                           const ushort_t* Wl, int sw,
                                           int nch, v8f acc[8], int lane) {
    const int nl = lane & 15;
    const int hi = lane >> 4;
    const ushort_t* arow = Astg + nl * sa + hi * 16;
    const ushort_t* wrow = Wl + nl * sw + hi * 16;
    for (int kc = 0; kc < nch; ++kc) {
        ABfrag a, b[8];
        const ushort_t* ap = arow + kc * 32;
        a.q[0] = *(const uint4*)ap;
        a.q[1] = *(const uint4*)(ap + 8);
#pragma unroll
        for (int nt = 0; nt < 8; ++nt) {
            const ushort_t* bp = wrow + nt * 16 * sw + kc * 32;
            b[nt].q[0] = *(const uint4*)bp;
            b[nt].q[1] = *(const uint4*)(bp + 8);
        }
        sched_fence();   // loads above / wmmas below: forces batched issue
#pragma unroll
        for (int nt = 0; nt < 8; ++nt)
            acc[nt] = __builtin_amdgcn_wmma_f32_16x16x32_bf16(
                false, a.v, false, b[nt].v, (short)0, acc[nt], false, false);
        sched_fence();   // keep next iteration's loads after this wmma batch
    }
}

// hidden-layer epilogue: bias + act, write bf16 h tile to LDS (stride 136),
// reset accumulators. C layout: m = r + (lane>>4)*8, n = nt*16 + (lane&15).
template <int ACT>
__device__ __forceinline__ void epilogue_h(v8f acc[8], const float* bias,
                                           ushort_t* stgH, int lane) {
    const int nl = lane & 15;
    const int mo = (lane >> 4) * 8;
#pragma unroll
    for (int nt = 0; nt < 8; ++nt) {
        float bb = __ldg(&bias[nt * 16 + nl]);
#pragma unroll
        for (int r = 0; r < 8; ++r) {
            float v = acc[nt][r] + bb;
            if (ACT == 1) v = fmaxf(v, 0.f);
            else          v = tanhf(v);
            stgH[(mo + r) * 136 + nt * 16 + nl] = f2bf(v);
            acc[nt][r] = 0.f;
        }
    }
}

// ---------------------------------------------------------------------------
// Fused 3-layer MLP.
// MODE: 0 = PLAIN (in0 row-major [rows,dinr], zero-padded to Kp0)
//       1 = CAT   (cols 0..127 <- in0[r], 128..255 <- in1[r])
//       2 = EDGE  (cols 0..127 <- in0[src[r]], 128..255 <- in0[dst[r]],
//                  256..383 <- in1[r]); src=ei[r], dst=ei[Etot+r]
// ACT: 0 = tanh, 1 = relu (hidden layers only; output layer linear)
// LN:  LayerNorm(out)*g+beta   D1: dout==1 (scalar tail, W2f f32)
// RES: out = resid + mlp(out)
// ---------------------------------------------------------------------------
template <int MODE, int ACT, bool LN, bool D1, bool RES>
__global__ __launch_bounds__(128) void mlp_fused(
    const ushort_t* __restrict__ Wt0, const ushort_t* __restrict__ Wt1,
    const ushort_t* __restrict__ Wt2,
    const float* __restrict__ b0, const float* __restrict__ b1,
    const float* __restrict__ b2,
    const float* __restrict__ gam, const float* __restrict__ bet,
    const float* __restrict__ W2f,
    const float* __restrict__ in0, const float* __restrict__ in1,
    const int* __restrict__ ei, const float* __restrict__ resid,
    float* __restrict__ out, int rows, int dinr, int Etot)
{
    extern __shared__ char smem[];
    const int tid  = threadIdx.x;
    const int lane = tid & 31;
    const int warp = tid >> 5;

    const int Kp0 = (dinr + 31) & ~31;
    const int SW0 = Kp0 + 8;                      // padded K-stride (halves)

    ushort_t* Wl0 = (ushort_t*)smem;
    ushort_t* Wl1 = Wl0 + 128 * SW0;
    ushort_t* Wl2 = Wl1 + 128 * 136;
    size_t woff = (size_t)128 * SW0 * 2 + (size_t)128 * 136 * 2 +
                  (D1 ? 0 : (size_t)128 * 136 * 2);
    size_t sstg = (size_t)16 * SW0 * 2;
    if (sstg < (size_t)16 * 136 * 2)   sstg = (size_t)16 * 136 * 2;
    if (sstg < (size_t)(16 * 132 + 32) * 4) sstg = (size_t)(16 * 132 + 32) * 4;
    sstg = (sstg + 15) & ~(size_t)15;

    char*     stgbase = smem + woff + (size_t)warp * sstg;
    ushort_t* stgA = (ushort_t*)stgbase;          // bf16 A tile (stride SW0)
    ushort_t* stgH = (ushort_t*)stgbase;          // bf16 h tile (stride 136)
    float*    stgF = (float*)stgbase;             // f32 z tile  (stride 132)

    // ---- block-cooperative weight load into LDS (column n per thread) ----
    {
        int n = tid;  // 128 threads
        for (int k = 0; k < Kp0; ++k)
            Wl0[n * SW0 + k] = (k < dinr) ? Wt0[(size_t)n * dinr + k] : (ushort_t)0;
        for (int k = 0; k < 128; ++k)
            Wl1[n * 136 + k] = Wt1[(size_t)n * 128 + k];
        if (!D1)
            for (int k = 0; k < 128; ++k)
                Wl2[n * 136 + k] = Wt2[(size_t)n * 128 + k];
    }
    __syncthreads();

    const int tile = blockIdx.x * 4 + warp;       // one 16-row tile per wave
    const int row0 = tile * 16;
    if (row0 >= rows) return;                     // wave-uniform skip

    // ---- stage A tile (gather + f32->bf16) --------------------------------
    if (MODE == 2) {
        for (int rr = 0; rr < 16; ++rr) {
            int r = row0 + rr;
            float4 v0 = {0, 0, 0, 0}, v1 = {0, 0, 0, 0}, v2 = {0, 0, 0, 0};
            if (r < rows) {
                int s = ei[r], d = ei[Etot + r];
                v0 = ((const float4*)(in0 + (size_t)s * 128))[lane];
                v1 = ((const float4*)(in0 + (size_t)d * 128))[lane];
                v2 = ((const float4*)(in1 + (size_t)r * 128))[lane];
            }
            ushort_t* dst = stgA + rr * SW0;
            store4bf(dst + 4 * lane, v0);
            store4bf(dst + 128 + 4 * lane, v1);
            store4bf(dst + 256 + 4 * lane, v2);
        }
    } else if (MODE == 1) {
        for (int rr = 0; rr < 16; ++rr) {
            int r = row0 + rr;
            float4 v0 = {0, 0, 0, 0}, v1 = {0, 0, 0, 0};
            if (r < rows) {
                v0 = ((const float4*)(in0 + (size_t)r * 128))[lane];
                v1 = ((const float4*)(in1 + (size_t)r * 128))[lane];
            }
            ushort_t* dst = stgA + rr * SW0;
            store4bf(dst + 4 * lane, v0);
            store4bf(dst + 128 + 4 * lane, v1);
        }
    } else {
        for (int rr = 0; rr < 16; ++rr) {
            int r = row0 + rr;
            for (int c = lane; c < Kp0; c += 32) {
                float v = (r < rows && c < dinr) ? in0[(size_t)r * dinr + c] : 0.f;
                stgA[rr * SW0 + c] = f2bf(v);
            }
        }
    }
    cfence();

    v8f zero8;
#pragma unroll
    for (int i = 0; i < 8; ++i) zero8[i] = 0.f;
    v8f acc[8];
#pragma unroll
    for (int t = 0; t < 8; ++t) acc[t] = zero8;

    // layer 0
    gemm_accum(stgA, SW0, Wl0, SW0, Kp0 / 32, acc, lane);
    cfence();
    epilogue_h<ACT>(acc, b0, stgH, lane);
    cfence();
    // layer 1
    gemm_accum(stgH, 136, Wl1, 136, 4, acc, lane);
    cfence();
    epilogue_h<ACT>(acc, b1, stgH, lane);
    cfence();

    if (D1) {
        // layer 2: [16,128] @ [128,1] + b2 (linear) -> out[row]
        if (lane < 16) {
            int row = row0 + lane;
            if (row < rows) {
                float s = __ldg(&b2[0]);
                for (int k = 0; k < 128; ++k)
                    s += bf2f(stgH[lane * 136 + k]) * __ldg(&W2f[k]);
                out[row] = s;
            }
        }
    } else {
        // layer 2 (linear) + LayerNorm + residual
        gemm_accum(stgH, 136, Wl2, 136, 4, acc, lane);
        cfence();
        const int nl = lane & 15;
        const int mo = (lane >> 4) * 8;
#pragma unroll
        for (int nt = 0; nt < 8; ++nt) {
            float bb = __ldg(&b2[nt * 16 + nl]);
#pragma unroll
            for (int r = 0; r < 8; ++r)
                stgF[(mo + r) * 132 + nt * 16 + nl] = acc[nt][r] + bb;
        }
        cfence();
        if (lane < 16) {                       // per-row mean / var
            float s = 0.f, s2 = 0.f;
            for (int c = 0; c < 128; ++c) {
                float v = stgF[lane * 132 + c];
                s += v; s2 += v * v;
            }
            float mu  = s * (1.f / 128.f);
            float var = s2 * (1.f / 128.f) - mu * mu;
            stgF[16 * 132 + lane]      = mu;
            stgF[16 * 132 + 16 + lane] = rsqrtf(var + 1e-5f);
        }
        cfence();
        float4 g4 = make_float4(1, 1, 1, 1), be4 = make_float4(0, 0, 0, 0);
        if (LN) {
            g4  = ((const float4*)gam)[lane];
            be4 = ((const float4*)bet)[lane];
        }
        for (int r = 0; r < 16; ++r) {
            int row = row0 + r;
            if (row >= rows) continue;
            float mu = stgF[16 * 132 + r];
            float rs = stgF[16 * 132 + 16 + r];
            float4 z = ((const float4*)(stgF + r * 132))[lane];
            float4 o;
            if (LN) {
                o.x = (z.x - mu) * rs * g4.x + be4.x;
                o.y = (z.y - mu) * rs * g4.y + be4.y;
                o.z = (z.z - mu) * rs * g4.z + be4.z;
                o.w = (z.w - mu) * rs * g4.w + be4.w;
            } else {
                o = z;
            }
            if (RES) {
                float4 rv = ((const float4*)(resid + (size_t)row * 128))[lane];
                o.x += rv.x; o.y += rv.y; o.z += rv.z; o.w += rv.w;
            }
            ((float4*)(out + (size_t)row * 128))[lane] = o;
        }
    }
}

// ---- weight transpose + f32->bf16 : W[S,K,128] -> Wt[S,128,K] -------------
__global__ void convt_kernel(const float* __restrict__ W,
                             ushort_t* __restrict__ out, int K, long total) {
    long idx = (long)blockIdx.x * blockDim.x + threadIdx.x;
    if (idx >= total) return;
    long s   = idx / ((long)K * 128);
    long rem = idx - s * (long)K * 128;
    int  k   = (int)(rem / 128);
    int  n   = (int)(rem - (long)k * 128);
    out[s * (long)128 * K + (long)n * K + k] = f2bf(W[idx]);
}

// ---- agg[d] += e[edge]  (f32 atomics; agg is L2-resident) -----------------
__global__ void scatter_add_kernel(const float* __restrict__ e,
                                   const int* __restrict__ dsti,
                                   float* __restrict__ agg, long Etot) {
    long idx = (long)blockIdx.x * blockDim.x + threadIdx.x;
    if (idx >= Etot * 32) return;
    long ed = idx >> 5;
    int  ln = (int)(idx & 31);
    int  d  = dsti[ed];
    float4 v = ((const float4*)(e + ed * 128))[ln];
    float* a = agg + (size_t)d * 128 + ln * 4;
    atomicAdd(a + 0, v.x); atomicAdd(a + 1, v.y);
    atomicAdd(a + 2, v.z); atomicAdd(a + 3, v.w);
}

// ---- L symmetrization + triangular mask + sqrt diag -----------------------
__global__ void post_edge_kernel(const float* __restrict__ Lraw,
                                 const int* __restrict__ rev,
                                 const int* __restrict__ ei,
                                 const float* __restrict__ eattr,
                                 float* __restrict__ Lfin, int Etot) {
    int i = blockIdx.x * blockDim.x + threadIdx.x;
    if (i >= Etot) return;
    int s = ei[i], d = ei[Etot + i];
    float v = 0.5f * (Lraw[i] + Lraw[rev[i]]);
    if (s < d)  v = 0.f;
    if (s == d) v = sqrtf(eattr[(size_t)i * 4 + 2] + eattr[(size_t)i * 4 + 3]);
    Lfin[i] = v;
}

__global__ void spmv_LTx_kernel(const float* __restrict__ Lfin,
                                const int* __restrict__ ei,
                                const float* __restrict__ ix,
                                float* __restrict__ LTx, int Etot) {
    int i = blockIdx.x * blockDim.x + threadIdx.x;
    if (i >= Etot) return;
    atomicAdd(&LTx[ei[Etot + i]], Lfin[i] * ix[ei[i]]);
}
__global__ void spmv_LLTx_kernel(const float* __restrict__ Lfin,
                                 const int* __restrict__ ei,
                                 const float* __restrict__ LTx,
                                 float* __restrict__ LLTx, int Etot) {
    int i = blockIdx.x * blockDim.x + threadIdx.x;
    if (i >= Etot) return;
    atomicAdd(&LLTx[ei[i]], Lfin[i] * LTx[ei[Etot + i]]);
}

// ---- pack tuple outputs flat: b | L | diag_ele | rf | edge_index | out_x --
__global__ void pack_kernel(float* __restrict__ out, long out_size, int N,
                            int Etot, const float* __restrict__ ix,
                            const float* __restrict__ nattr,
                            const float* __restrict__ LLTx,
                            const float* __restrict__ Lfin,
                            const float* __restrict__ eattr,
                            const int* __restrict__ slp,
                            const int* __restrict__ ei) {
    long idx = (long)blockIdx.x * blockDim.x + threadIdx.x;
    if (idx >= out_size) return;
    long p = idx;
    if (p < N) {
        out[idx] = (nattr[(size_t)p * 6 + 3] != 0.f) ? ix[p] : LLTx[p];
        return;
    }
    p -= N;
    if (p < Etot) { out[idx] = Lfin[p]; return; }
    p -= Etot;
    if (p < N) {
        int sp = slp[p];
        out[idx] = eattr[(size_t)sp * 4 + 2] + eattr[(size_t)sp * 4 + 3];
        return;
    }
    p -= N;
    if (p < 1) { out[idx] = 1.f; return; }     // reverse_factor
    p -= 1;
    if (p < 2L * Etot) { out[idx] = (float)ei[p]; return; }
    p -= 2L * Etot;
    out[idx] = 0.f;                            // output_x (zeros) + slack
}

// ---------------------------------------------------------------------------
static size_t calc_smem(int dinr, bool d1) {
    int Kp0 = (dinr + 31) & ~31;
    size_t SW0 = (size_t)Kp0 + 8;
    size_t s = 128 * SW0 * 2 + (size_t)128 * 136 * 2 +
               (d1 ? 0 : (size_t)128 * 136 * 2);
    size_t sstg = 16 * SW0 * 2;
    if (sstg < (size_t)16 * 136 * 2)        sstg = (size_t)16 * 136 * 2;
    if (sstg < (size_t)(16 * 132 + 32) * 4) sstg = (size_t)(16 * 132 + 32) * 4;
    sstg = (sstg + 15) & ~(size_t)15;
    return s + 4 * sstg;
}

extern "C" void kernel_launch(void* const* d_in, const int* in_sizes, int n_in,
                              void* d_out, int out_size, void* d_ws,
                              size_t ws_size, hipStream_t stream) {
    const int N = in_sizes[3];   // input_x
    const int E = in_sizes[5];   // rev_perm
    const int ITERS = 30;

    const float* node_attr = (const float*)d_in[0];
    const float* edge_attr = (const float*)d_in[1];
    const int*   ei        = (const int*)d_in[2];
    const float* input_x   = (const float*)d_in[3];
    const int*   rev_perm  = (const int*)d_in[5];
    const int*   slp       = (const int*)d_in[6];
    // params (setup_inputs insertion order, depth-first)
    const float* ne_W0 = (const float*)d_in[7];
    const float* ne_b0 = (const float*)d_in[8];
    const float* ne_W1 = (const float*)d_in[9];
    const float* ne_b1 = (const float*)d_in[10];
    const float* ne_W2 = (const float*)d_in[11];
    const float* ne_b2 = (const float*)d_in[12];
    const float* ne_g  = (const float*)d_in[13];
    const float* ne_be = (const float*)d_in[14];
    const float* ee_W0 = (const float*)d_in[15];
    const float* ee_b0 = (const float*)d_in[16];
    const float* ee_W1 = (const float*)d_in[17];
    const float* ee_b1 = (const float*)d_in[18];
    const float* ee_W2 = (const float*)d_in[19];
    const float* ee_b2 = (const float*)d_in[20];
    const float* ee_g  = (const float*)d_in[21];
    const float* ee_be = (const float*)d_in[22];
    const float* me_W0 = (const float*)d_in[23];
    const float* me_b0 = (const float*)d_in[24];
    const float* me_W1 = (const float*)d_in[25];
    const float* me_b1 = (const float*)d_in[26];
    const float* me_W2 = (const float*)d_in[27];
    const float* me_b2 = (const float*)d_in[28];
    const float* me_g  = (const float*)d_in[29];
    const float* me_be = (const float*)d_in[30];
    const float* mn_W0 = (const float*)d_in[31];
    const float* mn_b0 = (const float*)d_in[32];
    const float* mn_W1 = (const float*)d_in[33];
    const float* mn_b1 = (const float*)d_in[34];
    const float* mn_W2 = (const float*)d_in[35];
    const float* mn_b2 = (const float*)d_in[36];
    const float* mn_g  = (const float*)d_in[37];
    const float* mn_be = (const float*)d_in[38];
    const float* de_W0 = (const float*)d_in[39];
    const float* de_b0 = (const float*)d_in[40];
    const float* de_W1 = (const float*)d_in[41];
    const float* de_b1 = (const float*)d_in[42];
    const float* de_W2 = (const float*)d_in[43];  // [128,1] kept f32
    const float* de_b2 = (const float*)d_in[44];

    // ---- workspace carve ---------------------------------------------------
    char* base = (char*)d_ws;
    size_t off = 0;
    auto alloc = [&](size_t bytes) -> size_t {
        size_t o = off;
        off = (off + bytes + 255) & ~(size_t)255;
        return o;
    };
    size_t o_x    = alloc((size_t)N * 128 * 4);
    size_t o_agg  = alloc((size_t)N * 128 * 4);
    size_t o_e    = alloc((size_t)E * 128 * 4);
    size_t o_Lr   = alloc((size_t)E * 4);
    size_t o_Lf   = alloc((size_t)E * 4);
    size_t o_LTx  = alloc((size_t)N * 4);
    size_t o_LLTx = alloc((size_t)N * 4);
    size_t o_ne0 = alloc((size_t)6 * 128 * 2);
    size_t o_ne1 = alloc((size_t)128 * 128 * 2);
    size_t o_ne2 = alloc((size_t)128 * 128 * 2);
    size_t o_ee0 = alloc((size_t)4 * 128 * 2);
    size_t o_ee1 = alloc((size_t)128 * 128 * 2);
    size_t o_ee2 = alloc((size_t)128 * 128 * 2);
    size_t o_me0 = alloc((size_t)ITERS * 384 * 128 * 2);
    size_t o_me1 = alloc((size_t)ITERS * 128 * 128 * 2);
    size_t o_me2 = alloc((size_t)ITERS * 128 * 128 * 2);
    size_t o_mn0 = alloc((size_t)ITERS * 256 * 128 * 2);
    size_t o_mn1 = alloc((size_t)ITERS * 128 * 128 * 2);
    size_t o_mn2 = alloc((size_t)ITERS * 128 * 128 * 2);
    size_t o_de0 = alloc((size_t)128 * 128 * 2);
    size_t o_de1 = alloc((size_t)128 * 128 * 2);
    (void)ws_size; (void)n_in;

    float* x    = (float*)(base + o_x);
    float* agg  = (float*)(base + o_agg);
    float* e    = (float*)(base + o_e);
    float* Lraw = (float*)(base + o_Lr);
    float* Lfin = (float*)(base + o_Lf);
    float* LTx  = (float*)(base + o_LTx);
    float* LLTx = (float*)(base + o_LLTx);
    ushort_t* wt_ne0 = (ushort_t*)(base + o_ne0);
    ushort_t* wt_ne1 = (ushort_t*)(base + o_ne1);
    ushort_t* wt_ne2 = (ushort_t*)(base + o_ne2);
    ushort_t* wt_ee0 = (ushort_t*)(base + o_ee0);
    ushort_t* wt_ee1 = (ushort_t*)(base + o_ee1);
    ushort_t* wt_ee2 = (ushort_t*)(base + o_ee2);
    ushort_t* wt_me0 = (ushort_t*)(base + o_me0);
    ushort_t* wt_me1 = (ushort_t*)(base + o_me1);
    ushort_t* wt_me2 = (ushort_t*)(base + o_me2);
    ushort_t* wt_mn0 = (ushort_t*)(base + o_mn0);
    ushort_t* wt_mn1 = (ushort_t*)(base + o_mn1);
    ushort_t* wt_mn2 = (ushort_t*)(base + o_mn2);
    ushort_t* wt_de0 = (ushort_t*)(base + o_de0);
    ushort_t* wt_de1 = (ushort_t*)(base + o_de1);

    auto conv = [&](const float* W, ushort_t* outp, int K, long total) {
        int blk = (int)((total + 255) / 256);
        convt_kernel<<<blk, 256, 0, stream>>>(W, outp, K, total);
    };
    conv(ne_W0, wt_ne0, 6, 6L * 128);
    conv(ne_W1, wt_ne1, 128, 128L * 128);
    conv(ne_W2, wt_ne2, 128, 128L * 128);
    conv(ee_W0, wt_ee0, 4, 4L * 128);
    conv(ee_W1, wt_ee1, 128, 128L * 128);
    conv(ee_W2, wt_ee2, 128, 128L * 128);
    conv(me_W0, wt_me0, 384, (long)ITERS * 384 * 128);
    conv(me_W1, wt_me1, 128, (long)ITERS * 128 * 128);
    conv(me_W2, wt_me2, 128, (long)ITERS * 128 * 128);
    conv(mn_W0, wt_mn0, 256, (long)ITERS * 256 * 128);
    conv(mn_W1, wt_mn1, 128, (long)ITERS * 128 * 128);
    conv(mn_W2, wt_mn2, 128, (long)ITERS * 128 * 128);
    conv(de_W0, wt_de0, 128, 128L * 128);
    conv(de_W1, wt_de1, 128, 128L * 128);

    // fused-MLP instantiations + dynamic LDS budgets
    auto kEnc  = mlp_fused<0, 0, true,  false, false>;  // tanh + LN
    auto kEdge = mlp_fused<2, 1, true,  false, true>;   // relu + LN + resid
    auto kNode = mlp_fused<1, 1, true,  false, true>;
    auto kDec  = mlp_fused<0, 0, false, true,  false>;  // tanh hidden, dout=1
    size_t smN = calc_smem(6, false);
    size_t smE4 = calc_smem(4, false);
    size_t smMe = calc_smem(384, false);
    size_t smMn = calc_smem(256, false);
    size_t smDe = calc_smem(128, true);
    (void)hipFuncSetAttribute((const void*)kEnc,
        hipFuncAttributeMaxDynamicSharedMemorySize, (int)(smN > smE4 ? smN : smE4));
    (void)hipFuncSetAttribute((const void*)kEdge,
        hipFuncAttributeMaxDynamicSharedMemorySize, (int)smMe);
    (void)hipFuncSetAttribute((const void*)kNode,
        hipFuncAttributeMaxDynamicSharedMemorySize, (int)smMn);
    (void)hipFuncSetAttribute((const void*)kDec,
        hipFuncAttributeMaxDynamicSharedMemorySize, (int)smDe);

    auto grid_of = [](int rows) { return ((rows + 15) / 16 + 3) / 4; };

    // ---- encoders ----------------------------------------------------------
    kEnc<<<grid_of(N), 128, smN, stream>>>(
        wt_ne0, wt_ne1, wt_ne2, ne_b0, ne_b1, ne_b2, ne_g, ne_be, nullptr,
        node_attr, nullptr, nullptr, nullptr, x, N, 6, 0);
    kEnc<<<grid_of(E), 128, smE4, stream>>>(
        wt_ee0, wt_ee1, wt_ee2, ee_b0, ee_b1, ee_b2, ee_g, ee_be, nullptr,
        edge_attr, nullptr, nullptr, nullptr, e, E, 4, 0);

    // ---- 30 message-passing iterations ------------------------------------
    long sc_threads = (long)E * 32;
    int  sc_blocks  = (int)((sc_threads + 255) / 256);
    for (int it = 0; it < ITERS; ++it) {
        kEdge<<<grid_of(E), 128, smMe, stream>>>(
            wt_me0 + (size_t)it * 384 * 128,
            wt_me1 + (size_t)it * 128 * 128,
            wt_me2 + (size_t)it * 128 * 128,
            me_b0 + it * 128, me_b1 + it * 128, me_b2 + it * 128,
            me_g + it * 128, me_be + it * 128, nullptr,
            x, e, ei, e, e, E, 384, E);
        hipMemsetAsync(agg, 0, (size_t)N * 128 * 4, stream);
        scatter_add_kernel<<<sc_blocks, 256, 0, stream>>>(e, ei + E, agg, E);
        kNode<<<grid_of(N), 128, smMn, stream>>>(
            wt_mn0 + (size_t)it * 256 * 128,
            wt_mn1 + (size_t)it * 128 * 128,
            wt_mn2 + (size_t)it * 128 * 128,
            mn_b0 + it * 128, mn_b1 + it * 128, mn_b2 + it * 128,
            mn_g + it * 128, mn_be + it * 128, nullptr,
            x, agg, nullptr, x, x, N, 256, 0);
    }

    // ---- edge decoder ------------------------------------------------------
    kDec<<<grid_of(E), 128, smDe, stream>>>(
        wt_de0, wt_de1, wt_de1, de_b0, de_b1, de_b2, nullptr, nullptr, de_W2,
        e, nullptr, nullptr, nullptr, Lraw, E, 128, 0);

    // ---- postprocess -------------------------------------------------------
    int eb = (E + 255) / 256;
    post_edge_kernel<<<eb, 256, 0, stream>>>(Lraw, rev_perm, ei, edge_attr,
                                             Lfin, E);
    hipMemsetAsync(LTx, 0, (size_t)N * 4, stream);
    hipMemsetAsync(LLTx, 0, (size_t)N * 4, stream);
    spmv_LTx_kernel<<<eb, 256, 0, stream>>>(Lfin, ei, input_x, LTx, E);
    spmv_LLTx_kernel<<<eb, 256, 0, stream>>>(Lfin, ei, LTx, LLTx, E);

    long ob = ((long)out_size + 255) / 256;
    if (ob > 0)
        pack_kernel<<<(int)ob, 256, 0, stream>>>(
            (float*)d_out, (long)out_size, N, E, input_x, node_attr, LLTx,
            Lfin, edge_attr, slp, ei);
}
